// PointCompressor_30116310680161
// MI455X (gfx1250) — compile-verified, matrix operand order checked by reading, not run on hardware
//
#include <hip/hip_runtime.h>
#include <cstdint>
#include <cstddef>

// ---------------------------------------------------------------------------
// MI455X (gfx1250) implementation. All matmuls run through
// v_wmma_f32_16x16x32_f16 (wave32 WMMA, f32 accum). Activation tiles are
// double-buffered in LDS and streamed by the Tensor Data Mover
// (tensor_load_to_lds, s_wait_tensorcnt(1) pipelined); W tiles are staged as
// f16 in LDS; next K-slab is prefetched with global_prefetch_b8.
// ---------------------------------------------------------------------------

#define KNB 16   // neighbors
#define BM  64   // GEMM block tile rows
#define BNT 64   // GEMM block tile cols
#define BK  32   // GEMM K step (== WMMA K)

typedef __attribute__((ext_vector_type(16))) _Float16     v16h;
typedef __attribute__((ext_vector_type(8)))  float        v8f;
typedef __attribute__((ext_vector_type(4)))  unsigned int uint4v;
typedef __attribute__((ext_vector_type(8)))  int          int8v;
typedef __attribute__((ext_vector_type(4)))  int          int4v;

#ifndef __has_builtin
#define __has_builtin(x) 0
#endif
#if __has_builtin(__builtin_amdgcn_tensor_load_to_lds)
#define HAVE_TDM 1
#else
#define HAVE_TDM 0
#endif

// ------------------------------ KNN (top-16) --------------------------------
__global__ void knn16_kernel(const float* __restrict__ xyz, int n,
                             int* __restrict__ idx, float* __restrict__ dist) {
  int i = blockIdx.x * blockDim.x + threadIdx.x;
  if (i >= n) return;
  float xi = xyz[3 * i + 0], yi = xyz[3 * i + 1], zi = xyz[3 * i + 2];
  float bd[KNB];
  int   bi[KNB];
#pragma unroll
  for (int t = 0; t < KNB; ++t) { bd[t] = 3.4e38f; bi[t] = 0; }
  for (int j = 0; j < n; ++j) {
    float dx = xi - xyz[3 * j + 0];
    float dy = yi - xyz[3 * j + 1];
    float dz = zi - xyz[3 * j + 2];
    float d2 = dx * dx + dy * dy + dz * dz;
    if (d2 < bd[KNB - 1]) {
      int t = KNB - 1;
      while (t > 0 && bd[t - 1] > d2) {
        bd[t] = bd[t - 1]; bi[t] = bi[t - 1]; --t;
      }
      bd[t] = d2; bi[t] = j;
    }
  }
#pragma unroll
  for (int t = 0; t < KNB; ++t) {
    idx[(size_t)i * KNB + t]  = bi[t];
    dist[(size_t)i * KNB + t] = sqrtf(fmaxf(bd[t], 1e-12f));
  }
}

// ----------------------- relative-feature construction ----------------------
__global__ void relfeat_kernel(const float* __restrict__ xyz,
                               const int* __restrict__ idx,
                               const float* __restrict__ dist, int n,
                               float* __restrict__ rel) {
  int e = blockIdx.x * blockDim.x + threadIdx.x;
  if (e >= n * KNB) return;
  int i = e / KNB;
  int j = idx[e];
  float x  = xyz[3 * i + 0], y  = xyz[3 * i + 1], z  = xyz[3 * i + 2];
  float nx = xyz[3 * j + 0], ny = xyz[3 * j + 1], nz = xyz[3 * j + 2];
  float* r = rel + (size_t)e * 10;
  r[0] = dist[e];
  r[1] = x - nx; r[2] = y - ny; r[3] = z - nz;
  r[4] = x;      r[5] = y;      r[6] = z;
  r[7] = nx;     r[8] = ny;     r[9] = nz;
}

// ----------------------- TDM descriptor + issue -----------------------------
#if HAVE_TDM
__device__ __forceinline__ void tdm_issue(const float* X, int M, int K,
                                          int m0, int k0, float* ldsDst) {
  // 2D Tensor-DMA: tile = BM x BK f32, row stride = K elements.
  // tensor_dim0/1 bound the remaining tensor so OOB reads zero-fill LDS.
  uint64_t ga  = (uint64_t)(uintptr_t)(X + (size_t)m0 * K + k0);
  uint32_t la  = (uint32_t)(uintptr_t)ldsDst;
  uint32_t td0 = (uint32_t)(K - k0);
  uint32_t td1 = (uint32_t)(M - m0);
  uint4v g0 = { 1u,                                   // count=1, gather off
                la,                                   // lds_addr
                (uint32_t)ga,                         // global_addr[31:0]
                (uint32_t)((ga >> 32) & 0x01FFFFFFu)  // global_addr[56:32]
                    | 0x80000000u };                  // type=2 ("image")
  int8v g1 = { (int)(2u << 16),                           // data_size=4B
               (int)((td0 & 0xFFFFu) << 16),              // tensor_dim0 lo
               (int)((td0 >> 16) | ((td1 & 0xFFFFu) << 16)),
               (int)((td1 >> 16) | ((uint32_t)BK << 16)), // tile_dim0=BK
               (int)BM,                                   // tile_dim1=BM
               (int)(uint32_t)K,                          // dim0 stride
               0, 0 };
  int4v z4 = {0, 0, 0, 0};
  int8v z8 = {0, 0, 0, 0, 0, 0, 0, 0};
  __builtin_amdgcn_tensor_load_to_lds(g0, g1, z4, z4, z8, 0);
}
#endif

// --------------------------- WMMA GEMM (+bias) ------------------------------
// Y[M,N] = X[M,K] @ W[K,N] (+ bias). 8 wave32s per block; block tile 64x64.
// Each wave owns a 16x32 strip: one A fragment feeds two back-to-back WMMAs.
// A tiles stream via double-buffered TDM DMA; W tiles staged as f16 in LDS.
__global__ __launch_bounds__(256) void gemm_wmma_kernel(
    const float* __restrict__ X, const float* __restrict__ W,
    const float* __restrict__ bias, float* __restrict__ Y,
    int M, int K, int N) {
  __shared__ float    lx[2][BM][BK];  // ping-pong A tiles (raw f32 from TDM)
  __shared__ _Float16 lw[BK][BNT];    // W tile, pre-converted to f16
  const int tid   = threadIdx.x;
  const int wid   = tid >> 5;
  const int lane  = tid & 31;
  const int lmod  = lane & 15;
  const int lhalf = lane >> 4;
  const int m0 = blockIdx.y * BM;
  const int n0 = blockIdx.x * BNT;
  const int wm = (wid >> 1) << 4;  // 0,16,32,48
  const int wn = (wid & 1) << 5;   // 0,32
  v8f acc0 = {}, acc1 = {};
  const int nk = (K + BK - 1) / BK;
#if HAVE_TDM
  if (wid == 0) tdm_issue(X, M, K, m0, 0, &lx[0][0][0]);
#endif
  for (int it = 0; it < nk; ++it) {
    const int k0 = it * BK;
    __syncthreads();  // everyone done reading tiles of iteration it-1
#if HAVE_TDM
    if (wid == 0 && it + 1 < nk)
      tdm_issue(X, M, K, m0, k0 + BK, &lx[(it + 1) & 1][0][0]);
#else
    {
      float* dst = &lx[it & 1][0][0];
      for (int e = tid; e < BM * BK; e += 256) {
        int r = e >> 5, c = e & (BK - 1);
        int gm = m0 + r, gk = k0 + c;
        dst[e] = (gm < M && gk < K) ? X[(size_t)gm * K + gk] : 0.0f;
      }
    }
#endif
    // W tile staging (f32 -> f16), branch-free: clamped address + select.
    {
      const int r  = tid >> 3;        // 0..31  (K row within slab)
      const int cb = (tid & 7) << 3;  // 0,8,...,56
      const int gk = k0 + r;
      const int kr = (gk < K) ? gk : 0;
      const bool kv = (gk < K);
#pragma unroll
      for (int j = 0; j < 8; ++j) {
        const int gn = n0 + cb + j;
        const int cn = (gn < N) ? gn : 0;
        const float v = W[(size_t)kr * N + cn];
        lw[r][cb + j] = (_Float16)((kv && gn < N) ? v : 0.0f);
      }
    }
    if (it + 1 < nk)
      __builtin_prefetch(X + (size_t)m0 * K + (k0 + BK), 0, 1);
#if HAVE_TDM
    if (wid == 0) {
      if (it + 1 < nk) __builtin_amdgcn_s_wait_tensorcnt(1);  // slab it done
      else             __builtin_amdgcn_s_wait_tensorcnt(0);  // drain
    }
#endif
    __syncthreads();
    // Fragment build per ISA 16-bit A/B layout:
    //   lanes 0-15 : K = {0..7, 16..23};  lanes 16-31 : K = {8..15, 24..31}
    const float(*xa)[BK] = lx[it & 1];
    v16h a, b0, b1;
    const int arow = wm + lmod;
    const int kb   = lhalf << 3;
#pragma unroll
    for (int j = 0; j < 8; ++j) {
      a[j]      = (_Float16)xa[arow][kb + j];
      a[j + 8]  = (_Float16)xa[arow][16 + kb + j];
      b0[j]     = lw[kb + j][wn + lmod];
      b0[j + 8] = lw[16 + kb + j][wn + lmod];
      b1[j]     = lw[kb + j][wn + 16 + lmod];
      b1[j + 8] = lw[16 + kb + j][wn + 16 + lmod];
    }
    acc0 = __builtin_amdgcn_wmma_f32_16x16x32_f16(false, a, false, b0,
                                                  (short)0, acc0, false, false);
    acc1 = __builtin_amdgcn_wmma_f32_16x16x32_f16(false, a, false, b1,
                                                  (short)0, acc1, false, false);
  }
  // D layout: VGPR j -> row (j + 8*lhalf), col = lane%16
  const int col0 = n0 + wn + lmod;
  const int col1 = col0 + 16;
#pragma unroll
  for (int j = 0; j < 8; ++j) {
    const int row = m0 + wm + (lhalf << 3) + j;
    if (row < M) {
      if (col0 < N) {
        float v = acc0[j];
        if (bias) v += bias[col0];
        Y[(size_t)row * N + col0] = v;
      }
      if (col1 < N) {
        float v = acc1[j];
        if (bias) v += bias[col1];
        Y[(size_t)row * N + col1] = v;
      }
    }
  }
}

// ----------------------------- BatchNorm ------------------------------------
__global__ void bn_stats_kernel(const float* __restrict__ X, int rows, int C,
                                float* __restrict__ mean,
                                float* __restrict__ var) {
  int c = blockIdx.x;
  __shared__ float ss[256], sq[256];
  float s = 0.f, q = 0.f;
  for (int r = threadIdx.x; r < rows; r += 256) {
    float v = X[(size_t)r * C + c];
    s += v; q += v * v;
  }
  ss[threadIdx.x] = s; sq[threadIdx.x] = q;
  __syncthreads();
  for (int off = 128; off > 0; off >>= 1) {
    if (threadIdx.x < (unsigned)off) {
      ss[threadIdx.x] += ss[threadIdx.x + off];
      sq[threadIdx.x] += sq[threadIdx.x + off];
    }
    __syncthreads();
  }
  if (threadIdx.x == 0) {
    float m = ss[0] / (float)rows;
    mean[c] = m;
    var[c]  = fmaxf(sq[0] / (float)rows - m * m, 0.f);
  }
}

__global__ void bn_apply_kernel(const float* __restrict__ X,
                                const float* __restrict__ mean,
                                const float* __restrict__ var,
                                const float* __restrict__ g,
                                const float* __restrict__ b, int rows, int C,
                                int act, float* __restrict__ Y) {
  size_t e = (size_t)blockIdx.x * blockDim.x + threadIdx.x;
  if (e >= (size_t)rows * C) return;
  int c = (int)(e % C);
  float y = g[c] * (X[e] - mean[c]) * rsqrtf(var[c] + 1e-5f) + b[c];
  if (act) y = (y >= 0.f) ? y : 0.2f * y;
  Y[e] = y;
}

// ----------------------- gather + concat / softmax pool ---------------------
__global__ void gather_concat_kernel(const float* __restrict__ feat,
                                     const float* __restrict__ r,
                                     const int* __restrict__ idx, int n,
                                     int inc, int relc, float* __restrict__ f) {
  const int c = inc + relc;
  size_t e = (size_t)blockIdx.x * blockDim.x + threadIdx.x;
  size_t total = (size_t)n * KNB * c;
  if (e >= total) return;
  int ch = (int)(e % c);
  size_t ik = e / c;
  if (ch < inc) f[e] = feat[(size_t)idx[ik] * inc + ch];
  else          f[e] = r[ik * relc + (ch - inc)];
}

__global__ void softmax_pool_kernel(const float* __restrict__ L,
                                    const float* __restrict__ f, int n, int C,
                                    float* __restrict__ pooled) {
  size_t e = (size_t)blockIdx.x * blockDim.x + threadIdx.x;
  if (e >= (size_t)n * C) return;
  int c = (int)(e % C);
  int i = (int)(e / C);
  const float* lp = L + ((size_t)i * KNB) * C + c;
  const float* fp = f + ((size_t)i * KNB) * C + c;
  float mx = -3.4e38f;
  for (int kk = 0; kk < KNB; ++kk) mx = fmaxf(mx, lp[(size_t)kk * C]);
  float s = 0.f, acc = 0.f;
  for (int kk = 0; kk < KNB; ++kk) {
    float w = __expf(lp[(size_t)kk * C] - mx);
    s += w;
    acc += w * fp[(size_t)kk * C];
  }
  pooled[e] = acc / s;
}

// ----------------------------- elementwise ----------------------------------
__global__ void add_lrelu_kernel(const float* __restrict__ A,
                                 const float* __restrict__ B, size_t tot,
                                 float* __restrict__ Y) {
  size_t e = (size_t)blockIdx.x * blockDim.x + threadIdx.x;
  if (e >= tot) return;
  float v = A[e] + B[e];
  Y[e] = (v >= 0.f) ? v : 0.2f * v;
}

__global__ void add_kernel(const float* __restrict__ A,
                           const float* __restrict__ B, size_t tot,
                           float* __restrict__ Y) {
  size_t e = (size_t)blockIdx.x * blockDim.x + threadIdx.x;
  if (e >= tot) return;
  Y[e] = A[e] + B[e];
}

// ============================== host driver =================================
extern "C" void kernel_launch(void* const* d_in, const int* in_sizes, int n_in,
                              void* d_out, int out_size, void* d_ws,
                              size_t ws_size, hipStream_t stream) {
  (void)in_sizes; (void)n_in; (void)out_size; (void)ws_size;
  const float* fea    = (const float*)d_in[0];  // (1,8192,3) == xyz == feat0
  const float* qnoise = (const float*)d_in[1];  // (1,256,1024)

  int p = 2;
  auto nextp = [&]() -> const float* { return (const float*)d_in[p++]; };

  struct Lfa {
    const float *attn_w, *out_b, *out_be, *out_g, *out_w, *rel_b, *rel_be,
                *rel_g, *rel_w, *sc_b, *sc_be, *sc_g, *sc_w;
  };
  auto read_lfa = [&]() {
    Lfa l;
    l.attn_w = nextp(); l.out_b = nextp(); l.out_be = nextp();
    l.out_g  = nextp(); l.out_w = nextp(); l.rel_b  = nextp();
    l.rel_be = nextp(); l.rel_g = nextp(); l.rel_w  = nextp();
    l.sc_b   = nextp(); l.sc_be = nextp(); l.sc_g   = nextp();
    l.sc_w   = nextp();
    return l;
  };

  // jax pytree leaf order (dict keys sorted lexicographically)
  const float* dec0_b = nextp(); const float* dec0_be = nextp();
  const float* dec0_g = nextp(); const float* dec0_w  = nextp();
  const float* dec1_b = nextp(); const float* dec1_be = nextp();
  const float* dec1_g = nextp(); const float* dec1_w  = nextp();
  Lfa decg[2]; decg[0] = read_lfa(); decg[1] = read_lfa();
  const float* enc0_b = nextp(); const float* enc0_be = nextp();
  const float* enc0_g = nextp(); const float* enc0_w  = nextp();
  const float* enc1_b = nextp(); const float* enc1_be = nextp();
  const float* enc1_g = nextp(); const float* enc1_w  = nextp();

  static const int ENC_COUNT[5] = {6, 2, 2, 2, 2};
  static const int ENC_SPEC[5][6][3] = {
    {{3,16,24},{24,16,32},{32,16,48},{48,24,48},{48,24,64},{64,24,64}},
    {{64,24,128},{128,32,128},{0,0,0},{0,0,0},{0,0,0},{0,0,0}},
    {{128,32,256},{256,32,256},{0,0,0},{0,0,0},{0,0,0},{0,0,0}},
    {{256,32,512},{512,32,512},{0,0,0},{0,0,0},{0,0,0},{0,0,0}},
    {{512,64,1024},{1024,128,1024},{0,0,0},{0,0,0},{0,0,0},{0,0,0}},
  };
  Lfa encg[5][6];
  for (int g = 0; g < 5; ++g)
    for (int l = 0; l < ENC_COUNT[g]; ++l) encg[g][l] = read_lfa();

  // ---------------- workspace carve (floats) ----------------
  float* wsf = (float*)d_ws;
  size_t off = 0;
  auto carve = [&](size_t nfl) { float* q = wsf + off; off += nfl; return q; };
  float* relb  = carve((size_t)8192 * KNB * 10);
  int*   idxb  = (int*)carve((size_t)8192 * KNB);
  float* distb = carve((size_t)8192 * KNB);
  float* t1    = carve((size_t)8192 * KNB * 32);
  float* fbuf  = carve((size_t)12 * 1024 * 1024);
  float* Lbuf  = carve((size_t)12 * 1024 * 1024);
  float* pool  = carve((size_t)1024 * 1024);
  float* sbuf  = carve((size_t)1024 * 1024);
  float* obuf  = carve((size_t)1024 * 1024);
  float* fb0   = carve((size_t)1024 * 1024);
  float* fb1   = carve((size_t)1024 * 1024);
  float* meanb = carve(2048);
  float* varb  = carve(2048);
  float* fb[2] = {fb0, fb1};
  int cb = 0;
  auto outbuf = [&]() { float* t = fb[cb]; cb ^= 1; return t; };

  auto gemm = [&](const float* Xp, const float* Wp, const float* bp, float* Yp,
                  int M, int K, int N) {
    dim3 grid((unsigned)((N + BNT - 1) / BNT), (unsigned)((M + BM - 1) / BM));
    gemm_wmma_kernel<<<grid, dim3(256), 0, stream>>>(Xp, Wp, bp, Yp, M, K, N);
  };
  auto bn = [&](float* Xp, const float* gp, const float* bep, int rows, int C,
                int act, float* Yp) {
    bn_stats_kernel<<<dim3((unsigned)C), dim3(256), 0, stream>>>(Xp, rows, C,
                                                                 meanb, varb);
    size_t tot = (size_t)rows * C;
    bn_apply_kernel<<<dim3((unsigned)((tot + 255) / 256)), dim3(256), 0,
                      stream>>>(Xp, meanb, varb, gp, bep, rows, C, act, Yp);
  };
  auto knn = [&](int n) {
    knn16_kernel<<<dim3((unsigned)((n + 127) / 128)), dim3(128), 0, stream>>>(
        fea, n, idxb, distb);
    int tot = n * KNB;
    relfeat_kernel<<<dim3((unsigned)((tot + 255) / 256)), dim3(256), 0,
                     stream>>>(fea, idxb, distb, n, relb);
  };
  auto lfa = [&](int n, int inc, int relc, int outc, const Lfa& P,
                 const float* Fin) -> float* {
    const int c = inc + relc;
    const int M = n * KNB;
    gemm(relb, P.rel_w, P.rel_b, t1, M, 10, relc);
    bn(t1, P.rel_g, P.rel_be, M, relc, 1, t1);
    size_t ftot = (size_t)M * c;
    gather_concat_kernel<<<dim3((unsigned)((ftot + 255) / 256)), dim3(256), 0,
                           stream>>>(Fin, t1, idxb, n, inc, relc, fbuf);
    gemm(fbuf, P.attn_w, nullptr, Lbuf, M, c, c);
    size_t ptot = (size_t)n * c;
    softmax_pool_kernel<<<dim3((unsigned)((ptot + 255) / 256)), dim3(256), 0,
                          stream>>>(Lbuf, fbuf, n, c, pool);
    gemm(Fin, P.sc_w, P.sc_b, sbuf, n, inc, outc);
    bn(sbuf, P.sc_g, P.sc_be, n, outc, 0, sbuf);
    gemm(pool, P.out_w, P.out_b, obuf, n, c, outc);
    bn(obuf, P.out_g, P.out_be, n, outc, 0, obuf);
    float* T = outbuf();
    size_t etot = (size_t)n * outc;
    add_lrelu_kernel<<<dim3((unsigned)((etot + 255) / 256)), dim3(256), 0,
                       stream>>>(sbuf, obuf, etot, T);
    return T;
  };
  auto mlp = [&](const float* Xp, const float* Wp, const float* bp,
                 const float* gp, const float* bep, int rows, int K, int N,
                 int act) -> float* {
    float* T = outbuf();
    gemm(Xp, Wp, bp, T, rows, K, N);
    bn(T, gp, bep, rows, N, act, T);
    return T;
  };

  // ---------------- encoder ----------------
  const float* F = fea;
  int n = 8192;
  for (int g = 0; g < 5; ++g) {
    knn(n);
    for (int l = 0; l < ENC_COUNT[g]; ++l) {
      const int* sp = ENC_SPEC[g][l];
      F = lfa(n, sp[0], sp[1], sp[2], encg[g][l], F);
    }
    n >>= 1;  // truncation keeps leading rows; buffers stay contiguous
  }
  // n == 256 here
  F = mlp(F, enc0_w, enc0_b, enc0_g, enc0_be, n, 1024, 1024, 1);
  F = mlp(F, enc1_w, enc1_b, enc1_g, enc1_be, n, 1024, 1024, 0);
  {
    float* T = outbuf();
    size_t etot = (size_t)n * 1024;
    add_kernel<<<dim3((unsigned)((etot + 255) / 256)), dim3(256), 0, stream>>>(
        F, qnoise, etot, T);
    F = T;
  }
  // ---------------- decoder ----------------
  knn(n);
  F = lfa(n, 1024, 128, 1024, decg[0], F);
  F = lfa(n, 1024, 128, 1024, decg[1], F);
  F = mlp(F, dec0_w, dec0_b, dec0_g, dec0_be, n, 1024, 256, 1);
  {
    float* T = outbuf();
    gemm(F, dec1_w, dec1_b, T, n, 256, 96);
    bn(T, dec1_g, dec1_be, n, 96, 0, (float*)d_out);  // (256,96) -> out
  }
}
